// BinaryCascadeTransformer_19662360281628
// MI455X (gfx1250) — 8-bit matrix operand order repaired and checked by reading, compiled, not run on hardware
//
#include <hip/hip_runtime.h>
#include <hip/hip_bf16.h>

typedef int v8i __attribute__((ext_vector_type(8)));

#define BB 8
#define SS 1024
#define HH 1024
#define VV 32
#define MM (BB * SS)

__global__ void k_tern(const float* __restrict__ w, signed char* __restrict__ o, int n) {
  int i = blockIdx.x * blockDim.x + threadIdx.x;
  int stride = gridDim.x * blockDim.x;
  for (; i < n; i += stride) {
    float v = w[i];
    o[i] = (signed char)((v > 0.f) ? 1 : ((v < 0.f) ? -1 : 0));
  }
}

__global__ void k_rmsnorm(const float* __restrict__ x, float* __restrict__ y) {
  __shared__ float red[256];
  size_t row = blockIdx.x;
  const float* xr = x + row * HH;
  float s = 0.f;
  for (int i = threadIdx.x; i < HH; i += 256) s += xr[i];
  red[threadIdx.x] = s; __syncthreads();
  for (int o = 128; o > 0; o >>= 1) {
    if (threadIdx.x < o) red[threadIdx.x] += red[threadIdx.x + o];
    __syncthreads();
  }
  float mean = red[0] / (float)HH;
  __syncthreads();
  float v = 0.f;
  for (int i = threadIdx.x; i < HH; i += 256) { float d = xr[i] - mean; v += d * d; }
  red[threadIdx.x] = v; __syncthreads();
  for (int o = 128; o > 0; o >>= 1) {
    if (threadIdx.x < o) red[threadIdx.x] += red[threadIdx.x + o];
    __syncthreads();
  }
  float var = red[0] / (float)HH;
  float r = rsqrtf(fmaxf(var + 1e-8f, 1e-10f));
  float* yr = y + row * HH;
  for (int i = threadIdx.x; i < HH; i += 256) yr[i] = (xr[i] - mean) * r;
}

__global__ void k_quant_per_t(const float* __restrict__ xn, signed char* __restrict__ q8,
                              float* __restrict__ scales) {
  __shared__ float red[256];
  int t = blockIdx.x;
  float m = 0.f;
  for (int b = 0; b < BB; ++b) {
    const float* p = xn + ((size_t)b * SS + t) * HH;
    for (int i = threadIdx.x; i < HH; i += 256) m = fmaxf(m, fabsf(p[i]));
  }
  red[threadIdx.x] = m; __syncthreads();
  for (int o = 128; o > 0; o >>= 1) {
    if (threadIdx.x < o) red[threadIdx.x] = fmaxf(red[threadIdx.x], red[threadIdx.x + o]);
    __syncthreads();
  }
  float amax = red[0];
  float s = (amax == 0.f) ? 1.f : amax / 127.f;
  if (threadIdx.x == 0) scales[t] = s;
  float inv = 1.f / s;
  for (int b = 0; b < BB; ++b) {
    size_t base = ((size_t)b * SS + t) * HH;
    for (int i = threadIdx.x; i < HH; i += 256) {
      float qv = rintf(xn[base + i] * inv);
      qv = fminf(fmaxf(qv, -128.f), 127.f);
      q8[base + i] = (signed char)qv;
    }
  }
}

__global__ void k_zero_slots(float* p, int n) {
  if (blockIdx.x == 0 && (int)threadIdx.x < n) p[threadIdx.x] = 0.f;
}

__global__ void k_amax(const float* __restrict__ x, int n, float* __restrict__ amax) {
  __shared__ float red[256];
  int i = blockIdx.x * blockDim.x + threadIdx.x;
  int stride = gridDim.x * blockDim.x;
  float m = 0.f;
  for (; i < n; i += stride) m = fmaxf(m, fabsf(x[i]));
  red[threadIdx.x] = m; __syncthreads();
  for (int o = 128; o > 0; o >>= 1) {
    if (threadIdx.x < o) red[threadIdx.x] = fmaxf(red[threadIdx.x], red[threadIdx.x + o]);
    __syncthreads();
  }
  if (threadIdx.x == 0)
    atomicMax((unsigned int*)amax, __float_as_uint(red[0]));
}

__global__ void k_quant_global(const float* __restrict__ x, signed char* __restrict__ q8,
                               int n, const float* __restrict__ amax_p,
                               float* __restrict__ scale_out) {
  float amax = *amax_p;
  float s = (amax == 0.f) ? 1.f : amax / 127.f;
  if (blockIdx.x == 0 && threadIdx.x == 0) *scale_out = s;
  float inv = 1.f / s;
  int i = blockIdx.x * blockDim.x + threadIdx.x;
  int stride = gridDim.x * blockDim.x;
  for (; i < n; i += stride) {
    float qv = rintf(x[i] * inv);
    qv = fminf(fmaxf(qv, -128.f), 127.f);
    q8[i] = (signed char)qv;
  }
}

__device__ __forceinline__ v8i load_frag_A(const signed char* __restrict__ A, int lda, int lane) {
  int m = lane & 15, half = lane >> 4;
  const signed char* p = A + (size_t)m * lda + half * 16;
  int4 c0 = *(const int4*)(p + 0);
  int4 c1 = *(const int4*)(p + 32);
  v8i a;
  a[0] = c0.x; a[1] = c0.y; a[2] = c0.z; a[3] = c0.w;
  a[4] = c1.x; a[5] = c1.y; a[6] = c1.z; a[7] = c1.w;
  return a;
}

__device__ __forceinline__ v8i load_frag_B(const signed char* __restrict__ W, int ldw, int lane) {
  int n = lane & 15, khalf = lane >> 4;
  const signed char* p = W + (size_t)n * ldw + khalf * 16;
  int4 c0 = *(const int4*)(p + 0);
  int4 c1 = *(const int4*)(p + 32);
  v8i b;
  b[0] = c0.x; b[1] = c0.y; b[2] = c0.z; b[3] = c0.w;
  b[4] = c1.x; b[5] = c1.y; b[6] = c1.z; b[7] = c1.w;
  return b;
}

template <int NOUT>
__global__ void k_gemm_iu8(const signed char* __restrict__ Aq,
                           const signed char* __restrict__ W0,
                           const signed char* __restrict__ W1,
                           const signed char* __restrict__ W2,
                           const float* __restrict__ b0,
                           const float* __restrict__ b1,
                           const float* __restrict__ b2,
                           float* __restrict__ O0,
                           float* __restrict__ O1,
                           float* __restrict__ O2,
                           const float* __restrict__ scales, int smod,
                           int M, int N, int K) {
  int wave = blockIdx.x * (blockDim.x >> 5) + (threadIdx.x >> 5);
  int lane = threadIdx.x & 31;
  int ntiles_n = N >> 4;
  int tm = wave / ntiles_n;
  int tn = wave % ntiles_n;
  if (tm >= (M >> 4)) return;

  const signed char* Abase  = Aq + (size_t)(tm * 16) * K;
  const signed char* B0base = W0 + (size_t)(tn * 16) * K;
  const signed char* B1base = (NOUT > 1) ? W1 + (size_t)(tn * 16) * K : nullptr;
  const signed char* B2base = (NOUT > 2) ? W2 + (size_t)(tn * 16) * K : nullptr;

  v8i acc0 = {0, 0, 0, 0, 0, 0, 0, 0};
  v8i acc1 = {0, 0, 0, 0, 0, 0, 0, 0};
  v8i acc2 = {0, 0, 0, 0, 0, 0, 0, 0};

  for (int k0 = 0; k0 < K; k0 += 64) {
    v8i a = load_frag_A(Abase + k0, K, lane);
    v8i bf0 = load_frag_B(B0base + k0, K, lane);
    acc0 = __builtin_amdgcn_wmma_i32_16x16x64_iu8(true, a, true, bf0, acc0, false, false);
    if constexpr (NOUT > 1) {
      v8i bf1 = load_frag_B(B1base + k0, K, lane);
      acc1 = __builtin_amdgcn_wmma_i32_16x16x64_iu8(true, a, true, bf1, acc1, false, false);
    }
    if constexpr (NOUT > 2) {
      v8i bf2 = load_frag_B(B2base + k0, K, lane);
      acc2 = __builtin_amdgcn_wmma_i32_16x16x64_iu8(true, a, true, bf2, acc2, false, false);
    }
  }

  int ng = tn * 16 + (lane & 15);
  float bias0 = b0[ng];
  float bias1 = (NOUT > 1) ? b1[ng] : 0.f;
  float bias2 = (NOUT > 2) ? b2[ng] : 0.f;
  float sscal = smod ? 0.f : scales[0];

#pragma unroll
  for (int r = 0; r < 8; ++r) {
    int mg = tm * 16 + r + 8 * (lane >> 4);
    float s = smod ? scales[mg & (SS - 1)] : sscal;
    size_t idx = (size_t)mg * N + ng;
    O0[idx] = (float)acc0[r] * s + bias0;
    if constexpr (NOUT > 1) O1[idx] = (float)acc1[r] * s + bias1;
    if constexpr (NOUT > 2) O2[idx] = (float)acc2[r] * s + bias2;
  }
}

__global__ void k_scan(const float* __restrict__ F, const float* __restrict__ C,
                       const float* __restrict__ G, float* __restrict__ Y) {
  int idx = blockIdx.x * blockDim.x + threadIdx.x;
  if (idx >= BB * HH) return;
  int b = idx / HH, h = idx % HH;
  float hstate = 0.f;
  size_t base = (size_t)b * SS * HH + h;
  for (int t = 0; t < SS; ++t) {
    size_t i = base + (size_t)t * HH;
    float f = F[i], c = C[i], g = G[i];
    float sf = 1.f / (1.f + __expf(-f));
    float silu_c = c / (1.f + __expf(-c));
    hstate = sf * hstate + (1.f - sf) * silu_c;
    float sg = 1.f / (1.f + __expf(-g));
    Y[i] = hstate * sg;
  }
}

__global__ void k_silu_mul(const float* __restrict__ g, const float* __restrict__ u,
                           float* __restrict__ p, int n) {
  int i = blockIdx.x * blockDim.x + threadIdx.x;
  int stride = gridDim.x * blockDim.x;
  for (; i < n; i += stride) {
    float gv = g[i];
    p[i] = (gv / (1.f + __expf(-gv))) * u[i];
  }
}

__global__ void k_outproj(const float* __restrict__ D, const float* __restrict__ Wout,
                          const float* __restrict__ bout, float* __restrict__ out) {
  __shared__ float sh[8 * HH];
  int rb = blockIdx.x * 8;
  for (int i = threadIdx.x; i < 8 * HH; i += 256)
    sh[i] = D[(size_t)(rb + i / HH) * HH + (i % HH)];
  __syncthreads();
  int r = threadIdx.x >> 5;
  int v = threadIdx.x & 31;
  float acc = bout[v];
  const float* w = Wout + (size_t)v * HH;
  const float* dr = sh + r * HH;
  for (int hh = 0; hh < HH; ++hh) acc = fmaf(dr[hh], w[hh], acc);
  out[(size_t)(rb + r) * VV + v] = acc;
}

extern "C" void kernel_launch(void* const* d_in, const int* in_sizes, int n_in,
                              void* d_out, int out_size, void* d_ws, size_t ws_size,
                              hipStream_t stream) {
  const float* x     = (const float*)d_in[0];
  const float* W_f   = (const float*)d_in[1];
  const float* W_c   = (const float*)d_in[2];
  const float* W_g   = (const float*)d_in[3];
  const float* b_f   = (const float*)d_in[4];
  const float* b_c   = (const float*)d_in[5];
  const float* b_g   = (const float*)d_in[6];
  const float* Wg    = (const float*)d_in[7];
  const float* Wu    = (const float*)d_in[8];
  const float* Wd    = (const float*)d_in[9];
  const float* bg    = (const float*)d_in[10];
  const float* bu    = (const float*)d_in[11];
  const float* bd    = (const float*)d_in[12];
  const float* W_out = (const float*)d_in[13];
  const float* b_out = (const float*)d_in[14];
  float* out = (float*)d_out;

  const size_t HW = (size_t)HH * HH;
  const size_t MHf = (size_t)MM * HH * 4;

  char* ws = (char*)d_ws;
  signed char* tern = (signed char*)ws;
  signed char* q8   = (signed char*)(ws + 6 * HW);
  float* scales     = (float*)(ws + 6 * HW + (size_t)MM * HH);
  char* bufs        = ws + 6 * HW + (size_t)MM * HH + 4352;
  float* buf0 = (float*)(bufs + 0 * MHf);
  float* buf1 = (float*)(bufs + 1 * MHf);
  float* buf2 = (float*)(bufs + 2 * MHf);
  float* buf3 = (float*)(bufs + 3 * MHf);
  float* amaxZ  = scales + SS;
  float* scaleZ = scales + SS + 1;
  float* amaxP  = scales + SS + 2;
  float* scaleP = scales + SS + 3;

  const int GEMM_WAVES  = (MM / 16) * (HH / 16);
  const int GEMM_BLOCKS = GEMM_WAVES / 8;

  const float* Wsrc[6] = {W_f, W_c, W_g, Wg, Wu, Wd};
  for (int i = 0; i < 6; ++i)
    k_tern<<<4096, 256, 0, stream>>>(Wsrc[i], tern + (size_t)i * HW, (int)HW);

  k_rmsnorm<<<MM, 256, 0, stream>>>(x, buf0);
  k_quant_per_t<<<SS, 256, 0, stream>>>(buf0, q8, scales);

  k_gemm_iu8<3><<<GEMM_BLOCKS, 256, 0, stream>>>(
      q8, tern + 0 * HW, tern + 1 * HW, tern + 2 * HW,
      b_f, b_c, b_g, buf0, buf1, buf2, scales,  1, MM, HH, HH);

  k_scan<<<(BB * HH) / 256, 256, 0, stream>>>(buf0, buf1, buf2, buf3);

  k_rmsnorm<<<MM, 256, 0, stream>>>(buf3, buf0);
  k_zero_slots<<<1, 32, 0, stream>>>(amaxZ, 4);
  k_amax<<<1024, 256, 0, stream>>>(buf0, MM * HH, amaxZ);
  k_quant_global<<<4096, 256, 0, stream>>>(buf0, q8, MM * HH, amaxZ, scaleZ);

  k_gemm_iu8<2><<<GEMM_BLOCKS, 256, 0, stream>>>(
      q8, tern + 3 * HW, tern + 4 * HW, nullptr,
      bg, bu, nullptr, buf1, buf2, nullptr, scaleZ,  0, MM, HH, HH);

  k_silu_mul<<<4096, 256, 0, stream>>>(buf1, buf2, buf0, MM * HH);
  k_amax<<<1024, 256, 0, stream>>>(buf0, MM * HH, amaxP);
  k_quant_global<<<4096, 256, 0, stream>>>(buf0, q8, MM * HH, amaxP, scaleP);

  k_gemm_iu8<1><<<GEMM_BLOCKS, 256, 0, stream>>>(
      q8, tern + 5 * HW, nullptr, nullptr,
      bd, nullptr, nullptr, buf1, nullptr, nullptr, scaleP,  0, MM, HH, HH);

  k_outproj<<<MM / 8, 256, 0, stream>>>(buf1, W_out, b_out, out);

  (void)in_sizes; (void)n_in; (void)out_size; (void)ws_size;
}
